// AttentionBlock_49873160241751
// MI455X (gfx1250) — matrix operand order repaired and checked by reading, not compiled, not run on hardware
//
#include <hip/hip_runtime.h>

typedef __bf16 bf16;
typedef __attribute__((ext_vector_type(16))) __bf16 v16bf;
typedef __attribute__((ext_vector_type(8)))  __bf16 v8bf;
typedef __attribute__((ext_vector_type(4)))  __bf16 v4bf;
typedef __attribute__((ext_vector_type(8)))  float  v8f;
typedef __attribute__((ext_vector_type(4)))  int    v4i;

#define WMMA_BF16(a, b, c) \
  __builtin_amdgcn_wmma_f32_16x16x32_bf16(false, (a), false, (b), (short)0, (c), false, false)

// B=2, S=1024, HIDDEN=2880, N_HEADS=64, N_KV=8, D=64, QKV_DIM=5120, WINDOW=128, M=2048

#if defined(__has_builtin)
#if __has_builtin(__builtin_amdgcn_global_load_async_to_lds_b128)
#define ASYNC_CP_OK 1
#endif
#endif

#if defined(__has_builtin)
#if __has_builtin(__builtin_amdgcn_s_wait_asynccnt)
#define WAIT_ASYNC() __builtin_amdgcn_s_wait_asynccnt(0)
#else
#define WAIT_ASYNC() asm volatile("s_wait_asynccnt 0" ::: "memory")
#endif
#else
#define WAIT_ASYNC() asm volatile("s_wait_asynccnt 0" ::: "memory")
#endif

#ifdef ASYNC_CP_OK
typedef __attribute__((address_space(1))) v4i gas_v4i;   // global
typedef __attribute__((address_space(3))) v4i las_v4i;   // LDS
__device__ __forceinline__ void async_cp16B(const bf16* g, bf16* l) {
  // one 16B per-lane async copy: global -> LDS, tracked by ASYNCcnt
  __builtin_amdgcn_global_load_async_to_lds_b128(
      (gas_v4i*)g, (las_v4i*)l, 0, 0);
}
#endif

// DPP row_ror reduction within each 16-lane row (VALU pipe, no LDS traffic)
#define DPP_ROR_I(x, n) __builtin_amdgcn_update_dpp(0, (x), 0x120 + (n), 0xf, 0xf, true)
__device__ __forceinline__ float row16_max(float v) {
  v = fmaxf(v, __int_as_float(DPP_ROR_I(__float_as_int(v), 8)));
  v = fmaxf(v, __int_as_float(DPP_ROR_I(__float_as_int(v), 4)));
  v = fmaxf(v, __int_as_float(DPP_ROR_I(__float_as_int(v), 2)));
  v = fmaxf(v, __int_as_float(DPP_ROR_I(__float_as_int(v), 1)));
  return v;
}
__device__ __forceinline__ float row16_sum(float v) {
  v += __int_as_float(DPP_ROR_I(__float_as_int(v), 8));
  v += __int_as_float(DPP_ROR_I(__float_as_int(v), 4));
  v += __int_as_float(DPP_ROR_I(__float_as_int(v), 2));
  v += __int_as_float(DPP_ROR_I(__float_as_int(v), 1));
  return v;
}

__device__ __forceinline__ v16bf load_afrag(const bf16* p) {
  // A-frag (16x32 bf16): per-lane: 8 halves at K=kb.. and 8 halves at K=kb+16..
  v8bf lo = *(const v8bf*)p;
  v8bf hi = *(const v8bf*)(p + 16);
  return __builtin_shufflevector(lo, hi, 0,1,2,3,4,5,6,7,8,9,10,11,12,13,14,15);
}

// ---------------- f32 -> bf16 convert ----------------
__global__ __launch_bounds__(256) void cvt_bf16_kernel(const float* __restrict__ src,
                                                       bf16* __restrict__ dst, int n) {
  int i = (blockIdx.x * 256 + threadIdx.x) * 4;
  if (i < n) {
    float4 v = *(const float4*)(src + i);
    dst[i + 0] = (bf16)v.x; dst[i + 1] = (bf16)v.y;
    dst[i + 2] = (bf16)v.z; dst[i + 3] = (bf16)v.w;
  }
}

// ---------------- RMSNorm -> bf16 ----------------
__global__ __launch_bounds__(256) void rmsnorm_kernel(const float* __restrict__ x,
                                                      const float* __restrict__ w,
                                                      bf16* __restrict__ t) {
  const int row = blockIdx.x;           // 0..2047
  const int tid = threadIdx.x;
  const float* xr = x + (size_t)row * 2880;
  float ss = 0.f;
  for (int i = tid; i < 2880; i += 256) { float v = xr[i]; ss += v * v; }
  __shared__ float red[256];
  red[tid] = ss; __syncthreads();
  for (int off = 128; off > 0; off >>= 1) {
    if (tid < off) red[tid] += red[tid + off];
    __syncthreads();
  }
  float rs = rsqrtf(red[0] * (1.0f / 2880.0f) + 1e-5f);
  bf16* tr = t + (size_t)row * 2880;
  for (int i = tid; i < 2880; i += 256) tr[i] = (bf16)(xr[i] * rs * w[i]);
}

// ---------------- LDS-tiled WMMA GEMM ----------------
// C[M,N] = A[M,K] * W[N,K]^T + bias (+resid)
// block tile 128(M) x 128(N), BK=32, double-buffered LDS; 8 waves, wave tile 32x64
#define LDS_STRIDE 40   // 32 elems + 8 pad (80B rows -> conflict-free 16-lane frag reads)

template <bool HAS_RES>
__global__ __launch_bounds__(256) void gemm_bf16_kernel(const bf16* __restrict__ A,
                                                        const bf16* __restrict__ W,
                                                        const float* __restrict__ bias,
                                                        const float* __restrict__ resid,
                                                        float* __restrict__ C,
                                                        int M, int N, int K, int tiles_n) {
  __shared__ bf16 As[2][128 * LDS_STRIDE];
  __shared__ bf16 Bs[2][128 * LDS_STRIDE];

  const int tid  = threadIdx.x;
  const int lane = tid & 31;
  const int wid  = tid >> 5;
  const int wm   = wid >> 1;            // 0..3 (M direction)
  const int wn   = wid & 1;             // 0..1 (N direction)

  const int mt = blockIdx.x / tiles_n, nt = blockIdx.x % tiles_n;
  const int m0 = mt * 128, nB = nt * 128;

  // staging: each thread moves one 32B chunk of A and of B per K-step
  const int srow  = tid >> 1;           // 0..127
  const int shalf = (tid & 1) * 16;     // elem offset 0 / 16
  const int brow_g = (nB + srow < N) ? (nB + srow) : (N - 1);   // clamp tail tile
  const bf16* aptr = A + (size_t)(m0 + srow) * K + shalf;
  const bf16* bptr = W + (size_t)brow_g * K + shalf;
  bf16* asl[2] = { &As[0][srow * LDS_STRIDE + shalf], &As[1][srow * LDS_STRIDE + shalf] };
  bf16* bsl[2] = { &Bs[0][srow * LDS_STRIDE + shalf], &Bs[1][srow * LDS_STRIDE + shalf] };

  // fragment addressing within LDS tile
  const int akb = (lane >> 4) * 8;
  const int wkb = (lane >> 4) * 16;
  const int ar  = wm * 32 + (lane & 15);   // + i*16
  const int br  = wn * 64 + (lane & 15);   // + j*16

  v8f acc[2][4] = {};

  // prologue: stage k=0 into buffer 0
#ifdef ASYNC_CP_OK
  async_cp16B(aptr, asl[0]);     async_cp16B(aptr + 8, asl[0] + 8);
  async_cp16B(bptr, bsl[0]);     async_cp16B(bptr + 8, bsl[0] + 8);
  WAIT_ASYNC();
#else
  v16bf av = *(const v16bf*)(aptr);
  v16bf bv = *(const v16bf*)(bptr);
  *(v16bf*)asl[0] = av;
  *(v16bf*)bsl[0] = bv;
#endif
  __syncthreads();

  int buf = 0;
  for (int k = 0; k < K; k += 32) {
    const bool more = (k + 32) < K;
    if (more) {
#ifdef ASYNC_CP_OK
      async_cp16B(aptr + k + 32, asl[buf ^ 1]);
      async_cp16B(aptr + k + 40, asl[buf ^ 1] + 8);
      async_cp16B(bptr + k + 32, bsl[buf ^ 1]);
      async_cp16B(bptr + k + 40, bsl[buf ^ 1] + 8);
#else
      av = *(const v16bf*)(aptr + k + 32);
      bv = *(const v16bf*)(bptr + k + 32);
#endif
    }
    v16bf af[2], bfr[4];
#pragma unroll
    for (int i = 0; i < 2; ++i)
      af[i] = load_afrag(&As[buf][(ar + i * 16) * LDS_STRIDE + akb]);
#pragma unroll
    for (int j = 0; j < 4; ++j)
      bfr[j] = load_afrag(&Bs[buf][(br + j * 16) * LDS_STRIDE + akb]);
#pragma unroll
    for (int i = 0; i < 2; ++i)
#pragma unroll
      for (int j = 0; j < 4; ++j)
        acc[i][j] = WMMA_BF16(af[i], bfr[j], acc[i][j]);
    if (more) {
#ifdef ASYNC_CP_OK
      WAIT_ASYNC();
#else
      *(v16bf*)asl[buf ^ 1] = av;
      *(v16bf*)bsl[buf ^ 1] = bv;
#endif
      __syncthreads();
      buf ^= 1;
    }
  }

  // epilogue (branch-free residual via template)
  const int col = lane & 15, rbs = (lane >> 4) * 8;
#pragma unroll
  for (int i = 0; i < 2; ++i) {
#pragma unroll
    for (int j = 0; j < 4; ++j) {
      const int n = nB + wn * 64 + j * 16 + col;
      if (n < N) {
        const float bn = bias[n];
#pragma unroll
        for (int e = 0; e < 8; ++e) {
          const int m = m0 + wm * 32 + i * 16 + rbs + e;
          float v = acc[i][j][e] + bn;
          if (HAS_RES) v += resid[(size_t)m * N + n];
          C[(size_t)m * N + n] = v;
        }
      }
    }
  }
}

// ---------------- YaRN RoPE (+ split q/k/v to bf16) ----------------
__global__ __launch_bounds__(256) void rope_kernel(const float* __restrict__ qkv,
                                                   bf16* __restrict__ qo,
                                                   bf16* __restrict__ ko,
                                                   bf16* __restrict__ vo) {
  const int tid = blockIdx.x * 256 + threadIdx.x;   // total 2048*2560
  const int row = tid / 2560;
  const int u   = tid % 2560;
  if (row >= 2048) return;
  const int s = row & 1023;

  if (u < 2304) {
    int j; const float* src; bf16* dst;
    if (u < 2048) { int h = u >> 5; j = u & 31;
      src = qkv + (size_t)row * 5120 + h * 64;
      dst = qo  + (size_t)row * 4096 + h * 64;
    } else { int idx = u - 2048; int kv = idx >> 5; j = idx & 31;
      src = qkv + (size_t)row * 5120 + 4096 + kv * 64;
      dst = ko  + (size_t)row * 512 + kv * 64;
    }
    const float fj = (float)j;
    const float freq = powf(150000.0f, fj * (1.0f / 32.0f));
    const float conc = 0.1f * logf(32.0f) + 1.0f;
    const float log_base = logf(150000.0f);
    const float TWO_PI = 6.283185307f;
    const float low  = 32.0f * logf(4096.0f / (32.0f * TWO_PI)) / log_base;
    const float high = 32.0f * logf(4096.0f / (1.0f  * TWO_PI)) / log_base;
    float ramp = (fj - low) / (high - low);
    ramp = fminf(fmaxf(ramp, 0.0f), 1.0f);
    const float mk = 1.0f - ramp;
    const float inv_freq = (1.0f / (32.0f * freq)) * (1.0f - mk) + (1.0f / freq) * mk;
    const float ang = (float)s * inv_freq;
    const float c = cosf(ang) * conc, sn = sinf(ang) * conc;
    const float a = src[j], b = src[j + 32];
    dst[j]      = (bf16)(a * c - b * sn);
    dst[j + 32] = (bf16)(b * c + a * sn);
  } else {
    const int idx = u - 2304; const int kv = idx >> 5; const int j = idx & 31;
    const float* src = qkv + (size_t)row * 5120 + 4608 + kv * 64;
    bf16* dst = vo + (size_t)row * 512 + kv * 64;
    dst[j] = (bf16)src[j]; dst[j + 32] = (bf16)src[j + 32];
  }
}

// ---------------- flash attention, sliding window 128, sink logits ----------------
// one wave per (b, head, 16-query tile); block = 8 waves
__global__ __launch_bounds__(256) void attn_kernel(const bf16* __restrict__ qb_,
                                                   const bf16* __restrict__ kb_,
                                                   const bf16* __restrict__ vb_,
                                                   const float* __restrict__ sinks,
                                                   bf16* __restrict__ out) {
  __shared__ bf16 plds[8][256];                       // 16x16 P per wave
  __shared__ __attribute__((aligned(32))) bf16 vlds[8][1024];  // V^T [d][k] per wave

  const int lane = threadIdx.x & 31;
  const int wid  = threadIdx.x >> 5;
  const int gw   = blockIdx.x * 8 + wid;
  const int qt   = gw & 63;
  const int head = (gw >> 6) & 63;
  const int b    = gw >> 12;
  const int kv   = head >> 3;
  const int qbase = qt * 16;

  const int col = lane & 15;
  const int rb  = (lane >> 4) * 8;
  const int akb = (lane >> 4) * 8;
  const int dk  = (lane >> 4) * 16;

  // Q a-frags (16 rows x 64 dim -> 2 frags)
  const bf16* qrow = qb_ + ((size_t)(b * 1024 + qbase + (lane & 15)) * 4096) + head * 64;
  v16bf qa0 = load_afrag(qrow + akb);
  v16bf qa1 = load_afrag(qrow + 32 + akb);

  float mrow[8], lrow[8];
#pragma unroll
  for (int e = 0; e < 8; ++e) { mrow[e] = -INFINITY; lrow[e] = 0.f; }
  v8f o[4] = {};

  bf16* pw = &plds[wid][0];
  bf16* vw = &vlds[wid][0];
  const float SM_SCALE = 0.125f;   // 1/sqrt(64)
  const int kt_lo = (qt >= 8) ? (qt - 8) : 0;

  // V^T staging role: lane = (k-chunk of 4) x (d-chunk of 8) -> packed b64 LDS stores
  const int vkq = (lane & 3) * 4;    // k rows vkq..vkq+3
  const int vdq = (lane >> 2) * 8;   // d elems vdq..vdq+7

  for (int kt = kt_lo; kt <= qt; ++kt) {
    const int kbase = kt * 16;

    // K b-frags: lane=col holds key kbase+col, K-dim = head dim (contiguous)
    const bf16* krow = kb_ + ((size_t)(b * 1024 + kbase + col) * 512) + kv * 64;
    v16bf kf0 = load_afrag(krow + akb);
    v16bf kf1 = load_afrag(krow + 32 + akb);

    // stage V^T tile into LDS: vw[d*16 + k], 8x ds_store_b64 per lane
    {
      const bf16* vbase = vb_ + ((size_t)(b * 1024 + kbase + vkq) * 512) + kv * 64 + vdq;
      v8bf r0 = *(const v8bf*)(vbase);
      v8bf r1 = *(const v8bf*)(vbase + 512);
      v8bf r2 = *(const v8bf*)(vbase + 1024);
      v8bf r3 = *(const v8bf*)(vbase + 1536);
#pragma unroll
      for (int dd = 0; dd < 8; ++dd) {
        v4bf pk = { r0[dd], r1[dd], r2[dd], r3[dd] };
        *(v4bf*)&vw[(vdq + dd) * 16 + vkq] = pk;
      }
    }

    // S = Q K^T (16x16 f32)
    v8f sacc = {};
    sacc = WMMA_BF16(qa0, kf0, sacc);
    sacc = WMMA_BF16(qa1, kf1, sacc);

    // online softmax (rows live in 16-lane halves of the C layout; DPP reductions)
    float alpha_e[8];
#pragma unroll
    for (int e = 0; e < 8; ++e) {
      const int q = qbase + rb + e;
      const int k = kbase + col;
      float sv = sacc[e] * SM_SCALE;
      const bool valid = (k <= q) && (k > q - 128);
      sv = valid ? sv : -1e30f;
      const float rmax = row16_max(sv);
      const float mnew = fmaxf(mrow[e], rmax);
      const float p = __expf(sv - mnew);
      const float alpha = __expf(mrow[e] - mnew);
      const float psum = row16_sum(p);
      lrow[e] = lrow[e] * alpha + psum;
      mrow[e] = mnew;
      alpha_e[e] = alpha;
      pw[(rb + e) * 16 + col] = (bf16)p;
    }
#pragma unroll
    for (int dt = 0; dt < 4; ++dt)
#pragma unroll
      for (int e = 0; e < 8; ++e) o[dt][e] *= alpha_e[e];

    // wave-local LDS RAW fence (waves diverge in tile count; no block barrier)
    asm volatile("s_wait_dscnt 0" ::: "memory");

    // P a-frag (16x16 padded to 16x32 with zeros in upper K)
    v8bf plo = *(const v8bf*)(pw + (lane & 15) * 16 + akb);
    v8bf zz  = {};
    v16bf pa = __builtin_shufflevector(plo, zz, 0,1,2,3,4,5,6,7,8,9,10,11,12,13,14,15);

    // O += P * V   (P and V are both zero-padded in upper K)
#pragma unroll
    for (int dt = 0; dt < 4; ++dt) {
      v8bf vlo = *(const v8bf*)(vw + (dt * 16 + col) * 16 + akb);
      v16bf vf = __builtin_shufflevector(vlo, zz, 0,1,2,3,4,5,6,7,8,9,10,11,12,13,14,15);
      o[dt] = WMMA_BF16(pa, vf, o[dt]);
    }
  }

  // finalize with sink logit (adds only to the denominator)
  const float sk = sinks[head];
  float rr[8];
#pragma unroll
  for (int e = 0; e < 8; ++e) {
    const float mf = fmaxf(mrow[e], sk);
    const float sc = __expf(mrow[e] - mf);
    const float lf = lrow[e] * sc + __expf(sk - mf);
    rr[e] = sc / lf;
  }
#pragma unroll
  for (int dt = 0; dt < 4; ++dt) {
#pragma unroll
    for (int e = 0; e < 8; ++e) {
      const int q = qbase + rb + e;
      const int d = dt * 16 + col;
      out[((size_t)(b * 1024 + q) * 4096) + head * 64 + d] = (bf16)(o[dt][e] * rr[e]);
    }
  }
}

// ---------------- launch ----------------
extern "C" void kernel_launch(void* const* d_in, const int* in_sizes, int n_in,
                              void* d_out, int out_size, void* d_ws, size_t ws_size,
                              hipStream_t stream) {
  const float* x      = (const float*)d_in[0];
  const float* norm_w = (const float*)d_in[1];
  const float* qkv_w  = (const float*)d_in[2];
  const float* qkv_b  = (const float*)d_in[3];
  const float* out_w  = (const float*)d_in[4];
  const float* out_b  = (const float*)d_in[5];
  const float* sinks  = (const float*)d_in[6];
  float* out = (float*)d_out;

  char* ws = (char*)d_ws;
  bf16*  t_bf    = (bf16*)(ws);                        // 2048*2880 bf16
  bf16*  qkvw_bf = (bf16*)(ws + 11796480);             // 5120*2880 bf16
  bf16*  outw_bf = (bf16*)(ws + 41287680);             // 2880*4096 bf16
  float* qkv     = (float*)(ws + 64880640);            // 2048*5120 f32
  bf16*  q_bf    = (bf16*)(ws + 106823680);            // 2048*4096 bf16
  bf16*  k_bf    = (bf16*)(ws + 123600896);             // 2048*512 bf16
  bf16*  v_bf    = (bf16*)(ws + 125698048);             // 2048*512 bf16
  bf16*  attn_bf = (bf16*)(ws + 127795200);             // 2048*4096 bf16

  cvt_bf16_kernel<<<14400, 256, 0, stream>>>(qkv_w, qkvw_bf, 5120 * 2880);
  cvt_bf16_kernel<<<11520, 256, 0, stream>>>(out_w, outw_bf, 2880 * 4096);
  rmsnorm_kernel<<<2048, 256, 0, stream>>>(x, norm_w, t_bf);
  // qkv = t @ qkv_w^T + b : M=2048 N=5120 K=2880; tiles 16 x 40
  gemm_bf16_kernel<false><<<640, 256, 0, stream>>>(t_bf, qkvw_bf, qkv_b, nullptr, qkv,
                                                   2048, 5120, 2880, 40);
  rope_kernel<<<20480, 256, 0, stream>>>(qkv, q_bf, k_bf, v_bf);
  attn_kernel<<<1024, 256, 0, stream>>>(q_bf, k_bf, v_bf, sinks, attn_bf);
  // out = attn @ out_w^T + b + x : M=2048 N=2880 K=4096; tiles 16 x 23 (tail clamped)
  gemm_bf16_kernel<true><<<368, 256, 0, stream>>>(attn_bf, outw_bf, out_b, x, out,
                                                  2048, 2880, 4096, 23);
}
